// StandardMoE_88244398063761
// MI455X (gfx1250) — compile-verified, matrix operand order checked by reading, not run on hardware
//
#include <hip/hip_runtime.h>
#include <hip/hip_bf16.h>
#include <math.h>

#define D_MODEL 1024
#define D_FF    4096
#define N_EXP   8
#define T_TOK   8192   // B*S = 4*2048
#define TM      64     // token tile per block (4 m-tiles of 16)
#define MT      4      // TM/16
#define FC      256    // F chunk per outer iteration

typedef __bf16 bf16_t;
typedef __attribute__((ext_vector_type(16))) __bf16 v16bf;
typedef __attribute__((ext_vector_type(8)))  __bf16 v8bf;
typedef __attribute__((ext_vector_type(8)))  float  v8f;
typedef __attribute__((ext_vector_type(4)))  int    v4i;

#if defined(__has_builtin)
#if __has_builtin(__builtin_amdgcn_global_load_async_to_lds_b128)
#define HAVE_ASYNC_LDS 1
#endif
#endif

// Build a WMMA 16-bit operand from two contiguous 8-half (16B) chunks.
// Per CDNA5 ISA: lane L holds K = {g*16 + (L/16)*8 + j} for g=0,1.
__device__ __forceinline__ v16bf load_op(const bf16_t* p0, const bf16_t* p1) {
    v8bf lo = *reinterpret_cast<const v8bf*>(p0);
    v8bf hi = *reinterpret_cast<const v8bf*>(p1);
    v16bf r;
#pragma unroll
    for (int i = 0; i < 8; ++i) { r[i] = lo[i]; r[i + 8] = hi[i]; }
    return r;
}

__device__ __forceinline__ v8f wmma_bf16(v16bf a, v16bf b, v8f c) {
    return __builtin_amdgcn_wmma_f32_16x16x32_bf16(false, a, false, b,
                                                   (short)0, c, false, false);
}

// ---------------- transpose + fp32 -> bf16 cast -----------------------------
// src: [E][R][C] f32 -> dst: [E][C][R] bf16. Block = 256 threads (32x8 tile).
__global__ __launch_bounds__(256) void moe_transpose_cast_kernel(
    const float* __restrict__ src, bf16_t* __restrict__ dst, int R, int C) {
    __shared__ float tile[32][33];
    const size_t eoff = (size_t)blockIdx.z * R * C;
    const int c0 = blockIdx.x * 32, r0 = blockIdx.y * 32;
    const int tx = threadIdx.x & 31, ty = threadIdx.x >> 5;
#pragma unroll
    for (int i = 0; i < 4; ++i)
        tile[ty + i * 8][tx] = src[eoff + (size_t)(r0 + ty + i * 8) * C + c0 + tx];
    __syncthreads();
#pragma unroll
    for (int i = 0; i < 4; ++i)
        dst[eoff + (size_t)(c0 + ty + i * 8) * R + r0 + tx] =
            (bf16_t)tile[tx][ty + i * 8];
}

// ---------------- gating: softmax + top-2 + routing lists -------------------
// one wave32 per token; also casts x -> bf16
__global__ __launch_bounds__(256) void moe_gate_kernel(
    const float* __restrict__ x, const float* __restrict__ gw,
    bf16_t* __restrict__ xbf, int* __restrict__ cnt,
    int* __restrict__ toks, float* __restrict__ wgts) {
    const int wave = threadIdx.x >> 5, lane = threadIdx.x & 31;
    const int t = blockIdx.x * 8 + wave;
    const float* xr = x + (size_t)t * D_MODEL;

    float acc[N_EXP];
#pragma unroll
    for (int e = 0; e < N_EXP; ++e) acc[e] = 0.0f;

    for (int d = lane; d < D_MODEL; d += 32) {
        float xv = xr[d];
        xbf[(size_t)t * D_MODEL + d] = (bf16_t)xv;
#pragma unroll
        for (int e = 0; e < N_EXP; ++e)
            acc[e] += xv * gw[e * D_MODEL + d];
    }
#pragma unroll
    for (int e = 0; e < N_EXP; ++e) {
#pragma unroll
        for (int off = 16; off > 0; off >>= 1)
            acc[e] += __shfl_xor(acc[e], off, 32);
    }
    if (lane == 0) {
        float mx = acc[0];
#pragma unroll
        for (int e = 1; e < N_EXP; ++e) mx = fmaxf(mx, acc[e]);
        float p[N_EXP], s = 0.0f;
#pragma unroll
        for (int e = 0; e < N_EXP; ++e) { p[e] = __expf(acc[e] - mx); s += p[e]; }
        int i0 = 0; float b0 = p[0];
#pragma unroll
        for (int e = 1; e < N_EXP; ++e) if (p[e] > b0) { b0 = p[e]; i0 = e; }
        int i1 = (i0 == 0) ? 1 : 0; float b1 = p[i1];
#pragma unroll
        for (int e = 0; e < N_EXP; ++e)
            if (e != i0 && p[e] > b1) { b1 = p[e]; i1 = e; }
        float p0 = b0 / s, p1 = b1 / s;
        float inv = 1.0f / (p0 + p1 + 1e-9f);
        int pos0 = atomicAdd(&cnt[i0], 1);
        toks[i0 * T_TOK + pos0] = t; wgts[i0 * T_TOK + pos0] = p0 * inv;
        int pos1 = atomicAdd(&cnt[i1], 1);
        toks[i1 * T_TOK + pos1] = t; wgts[i1 * T_TOK + pos1] = p1 * inv;
    }
}

// ---------------- routed expert FFN: bf16 WMMA, fused GELU ------------------
// grid: (T_TOK/TM tiles, N_EXP experts); block = 256 threads = 8 waves.
// wave w owns output cols [w*128, w*128+128) and, in GEMM1, f-cols
// [fcb + w*32, fcb + w*32 + 32).
__global__ __launch_bounds__(256) void moe_ffn_kernel(
    const bf16_t* __restrict__ xbf, const bf16_t* __restrict__ w1t,
    const bf16_t* __restrict__ w2t, const float* __restrict__ gb1,
    const float* __restrict__ gb2, const int* __restrict__ cnt,
    const int* __restrict__ toks, const float* __restrict__ wgts,
    float* __restrict__ out) {
    __shared__ bf16_t xs[TM][D_MODEL];   // 128 KB
    __shared__ bf16_t hs[TM][FC];        // 32 KB
    __shared__ int    s_tok[TM];
    __shared__ float  s_w[TM];

    const int e = blockIdx.y;
    const int ne = cnt[e];
    const int m0 = blockIdx.x * TM;
    if (m0 >= ne) return;

    const int tid = threadIdx.x;
    const int wave = tid >> 5;
    const int lane = tid & 31;
    const int lhi = lane >> 4;   // 0/1 half-wave
    const int llo = lane & 15;

    if (tid < TM) {
        int i = m0 + tid;
        bool v = i < ne;
        s_tok[tid] = v ? toks[e * T_TOK + i] : 0;
        s_w[tid]   = v ? wgts[e * T_TOK + i] : 0.0f;
    }
    __syncthreads();

    // gather TM token rows of bf16 x into LDS (16B chunks, coalesced per row)
#if HAVE_ASYNC_LDS
    {
        typedef __attribute__((address_space(1))) v4i* gp_t;
        typedef __attribute__((address_space(3))) v4i* lp_t;
        for (int c = tid; c < TM * (D_MODEL / 8); c += 256) {
            int row = c >> 7;            // D_MODEL/8 = 128 chunks per row
            int off = (c & 127) << 3;
            __builtin_amdgcn_global_load_async_to_lds_b128(
                (gp_t)(xbf + (size_t)s_tok[row] * D_MODEL + off),
                (lp_t)&xs[row][off], 0, 0);
        }
#if __has_builtin(__builtin_amdgcn_s_wait_asynccnt)
        __builtin_amdgcn_s_wait_asynccnt(0);
#else
        asm volatile("s_wait_asynccnt 0x0" ::: "memory");
#endif
    }
#else
    for (int c = tid; c < TM * (D_MODEL / 8); c += 256) {
        int row = c >> 7;
        int off = (c & 127) << 3;
        *reinterpret_cast<v8bf*>(&xs[row][off]) =
            *reinterpret_cast<const v8bf*>(xbf + (size_t)s_tok[row] * D_MODEL + off);
    }
#endif
    __syncthreads();

    const bf16_t* w1e = w1t + (size_t)e * D_FF * D_MODEL;   // [F][D]
    const bf16_t* w2e = w2t + (size_t)e * D_MODEL * D_FF;   // [D][F]

    v8f acc[MT][8];
#pragma unroll
    for (int mt = 0; mt < MT; ++mt)
#pragma unroll
        for (int nt = 0; nt < 8; ++nt)
            acc[mt][nt] = (v8f){0.f,0.f,0.f,0.f,0.f,0.f,0.f,0.f};

    for (int fcb = 0; fcb < D_FF; fcb += FC) {
        // ---- GEMM1: h[TM, 32-per-wave] = x @ w1, K = D_MODEL ----
        v8f hacc[MT][2];
#pragma unroll
        for (int mt = 0; mt < MT; ++mt)
#pragma unroll
            for (int nt = 0; nt < 2; ++nt)
                hacc[mt][nt] = (v8f){0.f,0.f,0.f,0.f,0.f,0.f,0.f,0.f};

#pragma unroll 2
        for (int k = 0; k < D_MODEL / 32; ++k) {
            int d0 = k * 32 + lhi * 8;
            v16bf a[MT];
#pragma unroll
            for (int mt = 0; mt < MT; ++mt)
                a[mt] = load_op(&xs[mt * 16 + llo][d0], &xs[mt * 16 + llo][d0 + 16]);
#pragma unroll
            for (int nt = 0; nt < 2; ++nt) {
                int fcol = fcb + wave * 32 + nt * 16 + llo;
                const bf16_t* bp = w1e + (size_t)fcol * D_MODEL + d0;
                v16bf b = load_op(bp, bp + 16);
#pragma unroll
                for (int mt = 0; mt < MT; ++mt)
                    hacc[mt][nt] = wmma_bf16(a[mt], b, hacc[mt][nt]);
            }
        }
        __syncthreads();   // previous chunk's hs fully consumed

        // bias + exact GELU + bf16 -> hs
#pragma unroll
        for (int nt = 0; nt < 2; ++nt) {
            int col = wave * 32 + nt * 16 + llo;
            float bias = gb1[(size_t)e * D_FF + fcb + col];
#pragma unroll
            for (int mt = 0; mt < MT; ++mt) {
#pragma unroll
                for (int r = 0; r < 8; ++r) {
                    int row = mt * 16 + lhi * 8 + r;
                    float v = hacc[mt][nt][r] + bias;
                    float g = 0.5f * v * (1.0f + erff(v * 0.70710678118654752f));
                    hs[row][col] = (bf16_t)g;
                }
            }
        }
        __syncthreads();

        // ---- GEMM2: y[TM, 128-per-wave] += h @ w2, K = FC chunk ----
#pragma unroll 2
        for (int kk = 0; kk < FC / 32; ++kk) {
            int f0 = kk * 32 + lhi * 8;
            v16bf a[MT];
#pragma unroll
            for (int mt = 0; mt < MT; ++mt)
                a[mt] = load_op(&hs[mt * 16 + llo][f0], &hs[mt * 16 + llo][f0 + 16]);
#pragma unroll
            for (int nt = 0; nt < 8; ++nt) {
                int col = wave * 128 + nt * 16 + llo;
                const bf16_t* bp = w2e + (size_t)col * D_FF + fcb + f0;
                v16bf b = load_op(bp, bp + 16);
#pragma unroll
                for (int mt = 0; mt < MT; ++mt)
                    acc[mt][nt] = wmma_bf16(a[mt], b, acc[mt][nt]);
            }
        }
    }

    // ---- epilogue: + b2, * gate weight, scatter-add into out ----
#pragma unroll
    for (int nt = 0; nt < 8; ++nt) {
        int col = wave * 128 + nt * 16 + llo;
        float bias2 = gb2[(size_t)e * D_MODEL + col];
#pragma unroll
        for (int mt = 0; mt < MT; ++mt) {
#pragma unroll
            for (int r = 0; r < 8; ++r) {
                int m = mt * 16 + lhi * 8 + r;
                float val = (acc[mt][nt][r] + bias2) * s_w[m];
                atomicAdd(&out[(size_t)s_tok[m] * D_MODEL + col], val);
            }
        }
    }
}

// ---------------------------------------------------------------------------
extern "C" void kernel_launch(void* const* d_in, const int* in_sizes, int n_in,
                              void* d_out, int out_size, void* d_ws, size_t ws_size,
                              hipStream_t stream) {
    const float* x  = (const float*)d_in[0];   // [B,S,D]
    const float* gw = (const float*)d_in[1];   // [E,D]
    const float* w1 = (const float*)d_in[2];   // [E,D,F]
    const float* b1 = (const float*)d_in[3];   // [E,F]
    const float* w2 = (const float*)d_in[4];   // [E,F,D]
    const float* b2 = (const float*)d_in[5];   // [E,D]
    float* out = (float*)d_out;

    char* ws = (char*)d_ws;
    bf16_t* xbf = (bf16_t*)ws;  ws += (size_t)T_TOK * D_MODEL * sizeof(bf16_t);
    bf16_t* w1t = (bf16_t*)ws;  ws += (size_t)N_EXP * D_FF * D_MODEL * sizeof(bf16_t);
    bf16_t* w2t = (bf16_t*)ws;  ws += (size_t)N_EXP * D_MODEL * D_FF * sizeof(bf16_t);
    int*    cnt = (int*)ws;     ws += 256;
    int*    toks = (int*)ws;    ws += (size_t)N_EXP * T_TOK * sizeof(int);
    float*  wgts = (float*)ws;

    (void)hipMemsetAsync(cnt, 0, 256, stream);
    (void)hipMemsetAsync(d_out, 0, (size_t)out_size * sizeof(float), stream);

    dim3 blk(256);
    // w1 [E][D][F] -> w1t [E][F][D]
    moe_transpose_cast_kernel<<<dim3(D_FF / 32, D_MODEL / 32, N_EXP), blk, 0, stream>>>(
        w1, w1t, D_MODEL, D_FF);
    // w2 [E][F][D] -> w2t [E][D][F]
    moe_transpose_cast_kernel<<<dim3(D_MODEL / 32, D_FF / 32, N_EXP), blk, 0, stream>>>(
        w2, w2t, D_FF, D_MODEL);

    moe_gate_kernel<<<dim3(T_TOK / 8), blk, 0, stream>>>(x, gw, xbf, cnt, toks, wgts);

    moe_ffn_kernel<<<dim3(T_TOK / TM, N_EXP), blk, 0, stream>>>(
        xbf, w1t, w2t, b1, b2, cnt, toks, wgts, out);
}
